// MistralAttention_42640435315444
// MI455X (gfx1250) — matrix operand order repaired and checked by reading, not compiled, not run on hardware
//
#include <hip/hip_runtime.h>

// ---------------- problem constants ----------------
#define S_LEN 4096
#define NQ    32
#define NKVH  8
#define HD    128
#define MTOT  8192              // B * S
#define NTOT  6144              // NQ*HD + 2*NKVH*HD
#define KDIM  4096
#define BM    128
#define BN    128
#define BK    32
#define KT    (KDIM / BK)       // 128 k-steps
#define LDSW  40                // LDS row stride in halfwords (80 B, 16B-aligned, padded)
#define QELEMS  33554432ull     // 2*32*4096*128
#define KVELEMS 8388608ull      // 2*8*4096*128

typedef __attribute__((ext_vector_type(16))) __bf16        v16bf;
typedef __attribute__((ext_vector_type(8)))  float         v8f;
typedef __attribute__((ext_vector_type(4)))  float         v4f;
typedef __attribute__((ext_vector_type(4)))  unsigned int  v4u;

union Frag { v4u u[2]; v16bf v; };

// fp32 pair -> packed bf16 dword: round-half-up then a single v_perm_b32
// grabs the two high halfwords. (3 VALU ops total, adds can dual-issue.)
__device__ __forceinline__ unsigned int pk_bf16(float a, float b) {
    unsigned int ua = __float_as_uint(a) + 0x8000u;
    unsigned int ub = __float_as_uint(b) + 0x8000u;
#if __has_builtin(__builtin_amdgcn_perm)
    // {S0=ub bytes 4-7, S1=ua bytes 0-3}; take ua.b2,ua.b3,ub.b2,ub.b3
    return __builtin_amdgcn_perm(ub, ua, 0x07060302u);
#else
    return (ua >> 16) | (ub & 0xFFFF0000u);
#endif
}

// ---------------------------------------------------------------------------
// Fused QKV projection:  C(8192 x 6144) = X(8192x4096) * [qw;kw;vw]^T
// bf16 WMMA, f32 accumulate, double-buffered LDS staging.
// Wave tiling: 2 M-strips x 4 N-subtiles per wave (8 WMMAs / 12 ds_load_b128
// per K-step). Output scattered directly into (B, heads, S, HD) layout.
// ---------------------------------------------------------------------------
__global__ __launch_bounds__(256) void qkv_wmma_kernel(
    const float* __restrict__ x,
    const float* __restrict__ qw,
    const float* __restrict__ kw,
    const float* __restrict__ vw,
    float* __restrict__ out)
{
    __shared__ unsigned short sx[2][BM * LDSW];   // X tile  (bf16)
    __shared__ unsigned short swt[2][BN * LDSW];  // W tile  (bf16)

    const int bn = blockIdx.x;            // 0..47
    const int bm = blockIdx.y;            // 0..63
    const int n0 = bn * BN;
    const int m0 = bm * BM;

    // Per-block projection region (BN == HD, boundaries 128-aligned)
    const float* wmat;
    size_t       region_off;
    int          nheads, wr0;
    if (n0 < NQ * HD)              { wmat = qw; region_off = 0;                nheads = NQ;   wr0 = n0; }
    else if (n0 < (NQ + NKVH) * HD){ wmat = kw; region_off = QELEMS;           nheads = NKVH; wr0 = n0 - NQ * HD; }
    else                           { wmat = vw; region_off = QELEMS + KVELEMS; wr0 = n0 - (NQ + NKVH) * HD; nheads = NKVH; }
    const int head = wr0 >> 7;

    const int tid = threadIdx.x;
    const int ldr = tid >> 2;             // 0..63 : row within tile (2 passes of 64)
    const int ldc = (tid & 3) * 8;        // 0,8,16,24 : k-column start (8 floats)

    const float* xg = x    + (size_t)(m0  + ldr) * KDIM + ldc;
    const float* wg = wmat + (size_t)(wr0 + ldr) * KDIM + ldc;

    const int wave = tid >> 5;
    const int lane = tid & 31;
    const int half = lane >> 4;           // k-half selector (ISA fragment layout)
    const int r    = lane & 15;           // row/col within 16x16 subtile
    const int wm   = (wave & 3) * 32;     // this wave's M offset (2 strips of 16)
    const int wn   = (wave >> 2) * 64;    // this wave's N offset (4 subtiles of 16)

    v8f acc[2][4] = {};

    // ---- prologue: stage tile 0 into buffer 0
    #pragma unroll
    for (int pass = 0; pass < 2; ++pass) {
        const v4f* px = (const v4f*)(xg + (size_t)pass * 64 * KDIM);
        v4f a0 = px[0], a1 = px[1];
        unsigned int* dx = (unsigned int*)&sx[0][(ldr + pass * 64) * LDSW] + (ldc >> 1);
        dx[0] = pk_bf16(a0.x, a0.y); dx[1] = pk_bf16(a0.z, a0.w);
        dx[2] = pk_bf16(a1.x, a1.y); dx[3] = pk_bf16(a1.z, a1.w);
        const v4f* pw = (const v4f*)(wg + (size_t)pass * 64 * KDIM);
        v4f b0 = pw[0], b1 = pw[1];
        unsigned int* dw = (unsigned int*)&swt[0][(ldr + pass * 64) * LDSW] + (ldc >> 1);
        dw[0] = pk_bf16(b0.x, b0.y); dw[1] = pk_bf16(b0.z, b0.w);
        dw[2] = pk_bf16(b1.x, b1.y); dw[3] = pk_bf16(b1.z, b1.w);
    }
    __syncthreads();

    // ---- main K loop (double-buffered, one barrier per step)
    for (int kt = 0; kt < KT; ++kt) {
        const int cur = kt & 1;

        // issue global loads for next tile (overlap with WMMA below)
        v4f rx[4], rw[4];
        if (kt + 1 < KT) {
            const size_t ko = (size_t)(kt + 1) * BK;
            #pragma unroll
            for (int pass = 0; pass < 2; ++pass) {
                const v4f* px = (const v4f*)(xg + (size_t)pass * 64 * KDIM + ko);
                rx[pass * 2 + 0] = px[0];
                rx[pass * 2 + 1] = px[1];
                const v4f* pw = (const v4f*)(wg + (size_t)pass * 64 * KDIM + ko);
                rw[pass * 2 + 0] = pw[0];
                rw[pass * 2 + 1] = pw[1];
            }
        }
        if (kt + 2 < KT) {  // warm L2 for the tile after next
            __builtin_prefetch(xg + (size_t)(kt + 2) * BK, 0, 0);
            __builtin_prefetch(wg + (size_t)(kt + 2) * BK, 0, 0);
        }

        // ---- WMMA compute on current buffer
        {
            // A fragment (16-bit 16x32): lanes 0-15 hold K0-7 & K16-23,
            //                            lanes 16-31 hold K8-15 & K24-31
            Frag fa[2];
            #pragma unroll
            for (int st = 0; st < 2; ++st) {
                const v4u* ap = (const v4u*)&sx[cur][(wm + st * 16 + r) * LDSW];
                fa[st].u[0] = ap[half];
                fa[st].u[1] = ap[2 + half];
            }
            #pragma unroll
            for (int j = 0; j < 4; ++j) {
                // B fragment (32x16): lanes 0-15 hold K0-7 & K16-23, lanes 16-31 hold K8-15 & K24-31
                const v4u* bp = (const v4u*)&swt[cur][(wn + j * 16 + r) * LDSW];
                Frag fb;
                fb.u[0] = bp[half];
                fb.u[1] = bp[2 + half];
                #pragma unroll
                for (int st = 0; st < 2; ++st) {
                    acc[st][j] = __builtin_amdgcn_wmma_f32_16x16x32_bf16(
                        false, fa[st].v, false, fb.v, (short)0, acc[st][j], false, false);
                }
            }
        }

        // ---- stage next tile into other buffer
        if (kt + 1 < KT) {
            const int nxt = cur ^ 1;
            #pragma unroll
            for (int pass = 0; pass < 2; ++pass) {
                unsigned int* dx = (unsigned int*)&sx[nxt][(ldr + pass * 64) * LDSW] + (ldc >> 1);
                v4f a0 = rx[pass * 2], a1 = rx[pass * 2 + 1];
                dx[0] = pk_bf16(a0.x, a0.y); dx[1] = pk_bf16(a0.z, a0.w);
                dx[2] = pk_bf16(a1.x, a1.y); dx[3] = pk_bf16(a1.z, a1.w);
                unsigned int* dw = (unsigned int*)&swt[nxt][(ldr + pass * 64) * LDSW] + (ldc >> 1);
                v4f b0 = rw[pass * 2], b1 = rw[pass * 2 + 1];
                dw[0] = pk_bf16(b0.x, b0.y); dw[1] = pk_bf16(b0.z, b0.w);
                dw[2] = pk_bf16(b1.x, b1.y); dw[3] = pk_bf16(b1.z, b1.w);
            }
        }
        __syncthreads();
    }

    // ---- epilogue: scatter C tiles into (B, heads, S, HD) layout
    // A 128-row M block never crosses a batch boundary, so b / s bases are
    // loop-invariant; inner stores use one base pointer + immediate offsets.
    const int bb = m0 >> 12;                               // batch index (block constant)
    float* hbase = out + region_off
                 + (((size_t)(bb * nheads + head) * S_LEN) << 7);
    #pragma unroll
    for (int st = 0; st < 2; ++st) {
        const int s0 = (m0 & (S_LEN - 1)) + wm + st * 16 + half * 8;   // seq pos base
        #pragma unroll
        for (int j = 0; j < 4; ++j) {
            const int d = wn + j * 16 + r;                 // dim within head
            float* p = hbase + (((size_t)s0) << 7) + d;
            #pragma unroll
            for (int i = 0; i < 8; ++i) {
                p[i << 7] = acc[st][j][i];                 // imm-offset stores
            }
        }
    }
}

// ---------------------------------------------------------------------------
// In-place RoPE on q and k regions of d_out.
// cos/sin come from concat(freqs,freqs): table[d+64] == table[d], d in [0,64)
// ---------------------------------------------------------------------------
__global__ __launch_bounds__(256) void rope_kernel(
    float* __restrict__ out,
    const float* __restrict__ cost,
    const float* __restrict__ sint)
{
    const size_t QPAIRS = QELEMS >> 1;            // 16,777,216
    const size_t idx = (size_t)blockIdx.x * 256 + threadIdx.x;

    float* base;
    int    nhs;                                   // log2(heads)
    size_t p;
    if (idx < QPAIRS) { base = out;          nhs = 5; p = idx; }
    else              { base = out + QELEMS; nhs = 3; p = idx - QPAIRS; }

    const int    d  = (int)(p & 63);
    const size_t rh = p >> 6;                     // (b*nh + h)*S + s
    const int    s  = (int)(rh & (S_LEN - 1));
    const int    bb = (int)(p >> (18 + nhs));     // batch index (shift, no divide)
    const size_t rb = rh << 7;

    const float x1 = base[rb + d];
    const float x2 = base[rb + d + 64];
    const size_t ci = (((size_t)bb * S_LEN + s) << 7) + d;
    const float c  = cost[ci];
    const float sn = sint[ci];

    base[rb + d]      = x1 * c - x2 * sn;
    base[rb + d + 64] = x2 * c + x1 * sn;
}

// ---------------------------------------------------------------------------
extern "C" void kernel_launch(void* const* d_in, const int* in_sizes, int n_in,
                              void* d_out, int out_size, void* d_ws, size_t ws_size,
                              hipStream_t stream)
{
    const float* x    = (const float*)d_in[0];   // hidden_states (2,4096,4096)
    const float* cost = (const float*)d_in[1];   // cos (2,4096,128)
    const float* sint = (const float*)d_in[2];   // sin (2,4096,128)
    const float* qw   = (const float*)d_in[3];   // (4096,4096)
    const float* kw   = (const float*)d_in[4];   // (1024,4096)
    const float* vw   = (const float*)d_in[5];   // (1024,4096)
    float* out = (float*)d_out;

    dim3 grid(NTOT / BN, MTOT / BM);             // 48 x 64 blocks
    qkv_wmma_kernel<<<grid, 256, 0, stream>>>(x, qw, kw, vw, out);

    const size_t pairs = (QELEMS + KVELEMS) >> 1;     // 20,971,520
    rope_kernel<<<(unsigned)(pairs / 256), 256, 0, stream>>>(out, cost, sint);
}
